// HConvTranspose2d_44976897523747
// MI455X (gfx1250) — matrix operand order repaired and checked by reading, not compiled, not run on hardware
//
#include <hip/hip_runtime.h>
#include <hip/hip_bf16.h>
#include <math.h>

// ---------------- problem constants (match reference) ----------------
#define BATCH   8
#define CIN     256
#define HH      64
#define WW      64
#define COUT    128
#define KHW     9            // 3x3
#define MDIM    1152         // KHW * COUT
#define NPIX    32768        // BATCH*HH*WW
#define OHW     129
#define OPIX    16641        // 129*129
#define C_CURV  1.0f

typedef __attribute__((ext_vector_type(16))) __bf16 v16bf;
typedef __attribute__((ext_vector_type(8)))  __bf16 v8bf;
typedef __attribute__((ext_vector_type(8)))  float  v8f;
typedef __attribute__((ext_vector_type(4))) unsigned int u32x4;
typedef __attribute__((ext_vector_type(8))) int          i32x8;
typedef __attribute__((ext_vector_type(4))) int          i32x4;

// =====================================================================
// Kernel 1: pack x [B,CIN,H,W] f32 -> Xb [NPIX, CIN] bf16 (row-major),
//           and lam_fc[pix] = 2 / (1 - c*||x||^2).
// =====================================================================
__global__ __launch_bounds__(256) void pack_x_kernel(
    const float* __restrict__ x, __bf16* __restrict__ Xb,
    float* __restrict__ lam_fc) {
  __shared__ __align__(16) __bf16 Xs[64 * 256];
  __shared__ float ss[64];
  const int t  = threadIdx.x;
  const int p0 = blockIdx.x * 64;
  if (t < 64) ss[t] = 0.f;
  __syncthreads();

  const int pixL = t & 63;
  const int kg   = t >> 6;                 // 0..3 (64 channels each)
  const int pix  = p0 + pixL;
  const int b    = pix >> 12;              // / (H*W)
  const int hw   = pix & 4095;
  float acc = 0.f;
  const float* xp = x + (((size_t)b * CIN + kg * 64) << 12) + hw;
  #pragma unroll 4
  for (int kk = 0; kk < 64; ++kk) {
    float v = xp[(size_t)kk << 12];        // coalesced over hw
    acc += v * v;
    Xs[pixL * 256 + kg * 64 + kk] = (__bf16)v;
  }
  atomicAdd(&ss[pixL], acc);
  __syncthreads();

  const uint4* Xsv = reinterpret_cast<const uint4*>(Xs);
  uint4* Xbv = reinterpret_cast<uint4*>(Xb + (size_t)p0 * 256);
  #pragma unroll
  for (int it = 0; it < 8; ++it) Xbv[it * 256 + t] = Xsv[it * 256 + t];
  if (t < 64) lam_fc[p0 + t] = 2.0f / (1.0f - C_CURV * ss[t]);
}

// =====================================================================
// Kernel 2: pack z [CIN, M] f32 -> Zbt [M, CIN] bf16 (B^T layout),
//           plus per-column constants.
// =====================================================================
__global__ __launch_bounds__(256) void pack_z_kernel(
    const float* __restrict__ z, const float* __restrict__ r,
    __bf16* __restrict__ Zbt, float* __restrict__ c1,
    float* __restrict__ c2, float* __restrict__ c3) {
  __shared__ __align__(16) __bf16 Zs[32 * 256];
  __shared__ float zn2[32];
  const int t  = threadIdx.x;
  const int m0 = blockIdx.x * 32;
  if (t < 32) zn2[t] = 0.f;
  __syncthreads();

  const int mL = t & 31;
  const int kg = t >> 5;                   // 0..7 (32 k each)
  float acc = 0.f;
  #pragma unroll 4
  for (int kk = 0; kk < 32; ++kk) {
    int k = kg * 32 + kk;
    float v = z[(size_t)k * MDIM + m0 + mL];   // coalesced over m
    acc += v * v;
    Zs[mL * 256 + k] = (__bf16)v;
  }
  atomicAdd(&zn2[mL], acc);
  __syncthreads();

  const uint4* Zsv = reinterpret_cast<const uint4*>(Zs);
  uint4* Zv = reinterpret_cast<uint4*>(Zbt + (size_t)m0 * 256);
  #pragma unroll
  for (int it = 0; it < 4; ++it) Zv[it * 256 + t] = Zsv[it * 256 + t];
  if (t < 32) {
    int m = m0 + t;
    float zn  = fmaxf(sqrtf(zn2[t]), 1e-15f);
    float tcr = 2.0f * r[m];               // cs = sqrt(c) = 1
    c1[m] = coshf(tcr) / zn;
    c2[m] = sinhf(tcr);
    c3[m] = 2.0f * zn;
  }
}

// =====================================================================
// Kernel 3: WMMA GEMM (bf16 -> f32) fused with Poincare-FC epilogue.
// A-tile (64x256 bf16 = 32KB) is DMA'd into LDS by the Tensor Data Mover
// (TENSOR_LOAD_TO_LDS + s_wait_tensorcnt); 8 A fragments hoisted to VGPRs;
// B streamed from the L2-resident Zbt.
// =====================================================================
__global__ __launch_bounds__(256) void gemm_fc_kernel(
    const __bf16* __restrict__ Xb, const __bf16* __restrict__ Zbt,
    const float* __restrict__ lam_fc, const float* __restrict__ c1,
    const float* __restrict__ c2, const float* __restrict__ c3,
    float* __restrict__ vws, float* __restrict__ sws,
    float* __restrict__ lamf) {
  __shared__ __align__(16) __bf16 As[64 * 256];
  __shared__ float lamS[64];
  __shared__ float qS[64 * 9];
  __shared__ float sS[64];
  const int t  = threadIdx.x;
  const int p0 = blockIdx.x * 64;

  // ---- TDM path: one wave issues a 2D tensor DMA descriptor ----------
  if (t == 0) {
    const unsigned long long gaddr =
        (unsigned long long)(uintptr_t)(Xb + (size_t)p0 * 256);
    const unsigned int lds_off = (unsigned int)(uintptr_t)As;  // LDS byte offset
    // D# group 0: count=1 | lds_addr | global_addr[56:0] | type=2
    u32x4 g0;
    g0[0] = 1u;                                       // count=1, user mode
    g0[1] = lds_off;                                  // lds_addr
    g0[2] = (unsigned int)(gaddr & 0xffffffffu);      // global_addr[31:0] -> bits 95:64
    g0[3] = (unsigned int)((gaddr >> 32) & 0x1ffffffu) | (2u << 30);  // [120:96], type=2
    // D# group 1: data_size=1 (2B) | tensor_dim0=256 | tensor_dim1=64 |
    //             tile_dim0=256 | tile_dim1=64 | tensor_dim0_stride=256
    i32x8 g1;
    g1[0] = 0x00010000;        // workgroup_mask=0, data_size=1 (2 bytes)
    g1[1] = (int)(256u << 16); // tensor_dim0[15:0] in [31:16]
    g1[2] = (int)(64u << 16);  // tensor_dim0[31:16]=0, tensor_dim1[15:0]=64
    g1[3] = (int)(256u << 16); // tensor_dim1[31:16]=0, tile_dim0=256
    g1[4] = 64;                // tile_dim1=64, tile_dim2=0
    g1[5] = 256;               // tensor_dim0_stride[31:0]
    g1[6] = 0;                 // stride msbs / dim1_stride lsbs
    g1[7] = 0;
    i32x4 gz  = {0, 0, 0, 0};  // groups 2/3 unused (2D tensor)
    i32x8 gz8 = {0, 0, 0, 0, 0, 0, 0, 0};
    __builtin_amdgcn_tensor_load_to_lds(g0, g1, gz, gz, gz8, 0);
    __builtin_amdgcn_s_wait_tensorcnt(0);
  }
  if (t < 64) lamS[t] = lam_fc[p0 + t];
  for (int e = t; e < 64 * 9; e += 256) qS[e] = 0.f;
  __syncthreads();

  const int wv   = t >> 5;
  const int lane = t & 31;
  const int rg   = wv & 3;                 // pixel row-group
  const int cg   = wv >> 2;                // column-group (0/1)
  const int hi   = lane >> 4;              // half-wave select
  const int l15  = lane & 15;

  // Preload the 8 A fragments for this wave's 16 pixel rows (K = 8 x 32).
  // CDNA5 A layout: lanes 0-15 row M=l15 hold K {k0..k0+7, k0+16..k0+23};
  // lanes 16-31 hold K {k0+8..k0+15, k0+24..k0+31} of the same rows.
  v16bf afrag[8];
  {
    const __bf16* ap = As + (rg * 16 + l15) * 256 + hi * 8;
    #pragma unroll
    for (int kt = 0; kt < 8; ++kt) {
      const v8bf* p = reinterpret_cast<const v8bf*>(ap + kt * 32);
      v8bf lo = p[0];
      v8bf hh = p[2];  // +16 halfs
      afrag[kt] = __builtin_shufflevector(lo, hh,
          0, 1, 2, 3, 4, 5, 6, 7, 8, 9, 10, 11, 12, 13, 14, 15);
    }
  }

  for (int j = 0; j < 36; ++j) {
    const int n0 = (2 * j + cg) * 16;
    const int n  = n0 + l15;
    // B layout (matches A): lanes 0-15 col N=l15 hold K {k0..k0+7, k0+16..k0+23};
    // lanes 16-31 hold K {k0+8..k0+15, k0+24..k0+31} (Zbt is Z^T so K is contiguous per column).
    const __bf16* bp = Zbt + (size_t)n * 256 + hi * 8;
    if (j < 35) __builtin_prefetch(bp + 32 * 256, 0, 1);  // next tile of B
    v8f acc = {};
    #pragma unroll
    for (int kt = 0; kt < 8; ++kt) {
      const v8bf* p = reinterpret_cast<const v8bf*>(bp + kt * 32);
      v8bf lo = p[0];
      v8bf hh = p[2];
      v16bf bfrag = __builtin_shufflevector(lo, hh,
          0, 1, 2, 3, 4, 5, 6, 7, 8, 9, 10, 11, 12, 13, 14, 15);
      acc = __builtin_amdgcn_wmma_f32_16x16x32_bf16(
          false, afrag[kt], false, bfrag, (short)0, acc, false, false);
    }
    // Epilogue: v = c3 * asinh(c1*lam*xz - (lam-1)*c2); accumulate v^2
    const float c1v = c1[n], c2v = c2[n], c3v = c3[n];
    const int kk = n % 9;
    #pragma unroll
    for (int i = 0; i < 8; ++i) {
      const int pixL = rg * 16 + i + hi * 8;   // D layout: M = i + 8*hi
      const float lamv = lamS[pixL];
      const float a = c1v * lamv * acc[i] - (lamv - 1.0f) * c2v;
      const float v = c3v * asinhf(a);
      vws[(size_t)(p0 + pixL) * MDIM + n] = v;
      atomicAdd(&qS[pixL * 9 + kk], v * v);
    }
  }
  __syncthreads();

  if (t < 64) {
    float vn2 = 0.f;
    #pragma unroll
    for (int kk = 0; kk < 9; ++kk) vn2 += qS[t * 9 + kk];
    float s = 1.0f / (1.0f + sqrtf(1.0f + C_CURV * vn2));
    sws[p0 + t] = s;
    sS[t] = s;
  }
  __syncthreads();
  for (int e = t; e < 64 * 9; e += 256) {
    const int pixL = e / 9;
    const float s = sS[pixL];
    lamf[(size_t)p0 * 9 + e] = 2.0f / (1.0f - C_CURV * s * s * qS[e]);
  }
}

// =====================================================================
// Kernel 4: overlap-add scatter. One thread per (pixel, cout); 9 taps.
// num stored [b][opix][cout] so the 128 atomics per tap are contiguous.
// =====================================================================
__global__ __launch_bounds__(256) void fold_kernel(
    const float* __restrict__ vws, const float* __restrict__ sws,
    const float* __restrict__ lamf, float* __restrict__ num,
    float* __restrict__ den) {
  const int t    = threadIdx.x;
  const int pix  = blockIdx.x * 2 + (t >> 7);
  const int cout = t & 127;
  const float s  = sws[pix];
  const int b = pix >> 12, hw = pix & 4095;
  const int h = hw >> 6, w0 = hw & 63;
  const float* vp = vws + (size_t)pix * MDIM + cout * 9;
  const float* lp = lamf + (size_t)pix * 9;
  #pragma unroll
  for (int kk = 0; kk < 9; ++kk) {        // m = cout*9 + ki*3 + kj
    const float p   = s * vp[kk];
    const float lam = lp[kk];
    const int ki = kk / 3, kj = kk % 3;
    const int opix = (ki + 2 * h) * OHW + (kj + 2 * w0);
    atomicAdd(&num[(((size_t)b * OPIX + opix) << 7) + cout], lam * p);
    if (cout == 0)
      atomicAdd(&den[(size_t)b * OPIX + opix], lam - 1.0f);
  }
}

// =====================================================================
// Kernel 5: gyromidpoint finalize + Mobius half-scale; transposed
// coalesced writeout to [B][COUT][OH][OW].
// =====================================================================
__global__ __launch_bounds__(256) void finalize_kernel(
    const float* __restrict__ num, const float* __restrict__ den,
    float* __restrict__ out) {
  __shared__ float fS[32 * 128];
  __shared__ float fn2S[32];
  __shared__ float scS[32];
  const int t   = threadIdx.x;
  const int b   = blockIdx.y;
  const int op0 = blockIdx.x * 32;
  if (t < 32) fn2S[t] = 0.f;
  __syncthreads();

  #pragma unroll 2
  for (int it = 0; it < 16; ++it) {
    const int e = it * 256 + t;
    const int opL = e >> 7, cout = e & 127;
    const int opix = op0 + opL;
    float f = 0.f;
    if (opix < OPIX) {
      const float d = den[(size_t)b * OPIX + opix];
      f = num[(((size_t)b * OPIX + opix) << 7) + cout] / d;
    }
    fS[e] = f;
    float p2 = f * f;                      // wave reduce (all 32 lanes share opL)
    #pragma unroll
    for (int o = 16; o; o >>= 1) p2 += __shfl_xor(p2, o, 32);
    if ((t & 31) == 0) atomicAdd(&fn2S[opL], p2);
  }
  __syncthreads();
  if (t < 32) {
    const float fn  = sqrtf(fn2S[t]);
    const float arg = fminf(fn, 1.0f - 1e-7f);
    scS[t] = tanhf(0.5f * atanhf(arg)) / fmaxf(fn, 1e-15f);
  }
  __syncthreads();
  #pragma unroll 2
  for (int it = 0; it < 16; ++it) {
    const int e = it * 256 + t;
    const int opL = e & 31, cout = e >> 5;
    const int opix = op0 + opL;
    if (opix < OPIX)
      out[((size_t)b * COUT + cout) * OPIX + opix] = fS[opL * 128 + cout] * scS[opL];
  }
}

// =====================================================================
extern "C" void kernel_launch(void* const* d_in, const int* in_sizes, int n_in,
                              void* d_out, int out_size, void* d_ws, size_t ws_size,
                              hipStream_t stream) {
  const float* x = (const float*)d_in[0];   // [8,256,64,64]
  const float* z = (const float*)d_in[1];   // [256,1152]
  const float* r = (const float*)d_in[2];   // [1152]
  float* out = (float*)d_out;               // [8,128,129,129]

  char* ws = (char*)d_ws;
  size_t off = 0;
  auto carve = [&](size_t bytes) {
    void* p = ws + off;
    off = (off + bytes + 255) & ~(size_t)255;
    return p;
  };
  __bf16* Xb     = (__bf16*)carve((size_t)NPIX * CIN * 2);
  __bf16* Zbt    = (__bf16*)carve((size_t)MDIM * CIN * 2);
  float*  lam_fc = (float*)carve((size_t)NPIX * 4);
  float*  c1     = (float*)carve((size_t)MDIM * 4);
  float*  c2     = (float*)carve((size_t)MDIM * 4);
  float*  c3     = (float*)carve((size_t)MDIM * 4);
  float*  vws    = (float*)carve((size_t)NPIX * MDIM * 4);
  float*  sws    = (float*)carve((size_t)NPIX * 4);
  float*  lamf   = (float*)carve((size_t)NPIX * KHW * 4);
  float*  num    = (float*)carve((size_t)BATCH * OPIX * COUT * 4);
  float*  den    = (float*)carve((size_t)BATCH * OPIX * 4);

  (void)hipMemsetAsync(num, 0, (size_t)BATCH * OPIX * COUT * 4, stream);
  (void)hipMemsetAsync(den, 0, (size_t)BATCH * OPIX * 4, stream);

  pack_x_kernel<<<NPIX / 64, 256, 0, stream>>>(x, Xb, lam_fc);
  pack_z_kernel<<<MDIM / 32, 256, 0, stream>>>(z, r, Zbt, c1, c2, c3);
  gemm_fc_kernel<<<NPIX / 64, 256, 0, stream>>>(Xb, Zbt, lam_fc, c1, c2, c3,
                                                vws, sws, lamf);
  fold_kernel<<<NPIX / 2, 256, 0, stream>>>(vws, sws, lamf, num, den);
  finalize_kernel<<<dim3((OPIX + 31) / 32, BATCH), 256, 0, stream>>>(num, den, out);
}
